// PointTransformer_34127810134633
// MI455X (gfx1250) — compile-verified, matrix operand order checked by reading, not run on hardware
//
#include <hip/hip_runtime.h>
#include <hip/hip_bf16.h>

// ---------------------------------------------------------------------------
// Types for CDNA5 WMMA
// ---------------------------------------------------------------------------
typedef __attribute__((ext_vector_type(16))) __bf16 bf16x16;
typedef __attribute__((ext_vector_type(8)))  float  floatx8;

union FragAB { bf16x16 v; uint4 u[2]; };

__device__ __forceinline__ floatx8 zero8() {
  floatx8 r;
#pragma unroll
  for (int i = 0; i < 8; ++i) r[i] = 0.0f;
  return r;
}

// CDNA5 async Global -> LDS copy (ASYNCcnt-tracked), 16 bytes per lane.
__device__ __forceinline__ void async_ld16(void* lds, const void* gaddr) {
  unsigned lds_off = (unsigned)(unsigned long long)lds;   // low 32 bits = LDS offset
  unsigned long long ga = (unsigned long long)gaddr;
  asm volatile("global_load_async_to_lds_b128 %0, %1, off"
               :: "v"(lds_off), "v"(ga) : "memory");
}
__device__ __forceinline__ void wait_async0() {
  asm volatile("s_wait_asynccnt 0" ::: "memory");
}

// Dimensions (compile-time constants of the problem)
#define Bn    32
#define Gn    512
#define Cn    768
#define Sn    513          // G+1
#define M0n   (Bn * Sn)    // 16416
#define BGn   (Bn * Gn)    // 16384
#define Hn    12
#define HDn   64
#define FFn   3072
#define ADn   16
#define NCLn  2048
#define D1n   6272         // B * GRID(196)
#define NVn   6
#define N1n   128
#define MIDn  12
#define H1n   6

// GEMM epilogue flags
#define EP_F32  1
#define EP_BF16 2
#define EP_GELU 4
#define EP_RES  8

// ---------------------------------------------------------------------------
// fp32 -> bf16 convert
// ---------------------------------------------------------------------------
__global__ void k_cvt_bf16(const float* __restrict__ src, __bf16* __restrict__ dst, size_t n) {
  size_t i = (size_t)blockIdx.x * blockDim.x + threadIdx.x;
  size_t stride = (size_t)gridDim.x * blockDim.x;
  for (; i < n; i += stride) dst[i] = (__bf16)src[i];
}

#define TM 128
#define TN 128
#define TK 32

// ---------------------------------------------------------------------------
// Shared GEMM epilogue
// ---------------------------------------------------------------------------
__device__ __forceinline__ void gemm_epilogue(
    floatx8 (&acc)[4][2], const float* bias, const float* res,
    float* outF, __bf16* outB, int M, int N,
    int m0, int n0, int wm, int wn, int lm, int half, int flags)
{
#pragma unroll
  for (int r = 0; r < 4; ++r)
#pragma unroll
    for (int c = 0; c < 2; ++c)
#pragma unroll
      for (int v = 0; v < 8; ++v) {
        int row = m0 + wm * 64 + r * 16 + half * 8 + v;
        int col = n0 + wn * 32 + c * 16 + lm;
        if (row < M && col < N) {
          float x = acc[r][c][v] + (bias ? bias[col] : 0.0f);
          if (flags & EP_RES)  x += res[(size_t)row * N + col];
          if (flags & EP_GELU) x = x / (1.0f + __expf(-1.702f * x));  // quick_gelu
          if (flags & EP_F32)  outF[(size_t)row * N + col] = x;
          if (flags & EP_BF16) outB[(size_t)row * N + col] = (__bf16)x;
        }
      }
}

// ---------------------------------------------------------------------------
// Fast WMMA GEMM (K % 32 == 0): async Global->LDS staging, double buffered.
//   out = epi(A @ W^T + bias [+ res]);  A: MxK bf16, W: NxK bf16.
//   Block tile 128x128, K-step 32, 8 waves (2x4), wave tile 64x32.
// ---------------------------------------------------------------------------
__global__ __launch_bounds__(256) void k_gemm_fast(
    const __bf16* __restrict__ A, const __bf16* __restrict__ W,
    const float* __restrict__ bias, const float* __restrict__ res,
    float* __restrict__ outF, __bf16* __restrict__ outB,
    int M, int N, int K, int flags)
{
  __shared__ alignas(16) __bf16 As[2][TM * TK];
  __shared__ alignas(16) __bf16 Bs[2][TN * TK];

  const int m0 = blockIdx.x * TM;
  const int n0 = blockIdx.y * TN;
  const int t  = threadIdx.x;
  const int l  = t & 31;
  const int w  = t >> 5;
  const int wm = w >> 2;         // 0..1
  const int wn = w & 3;          // 0..3
  const int lm   = l & 15;
  const int half = l >> 4;
  const int koff = half * 8;

  // staging coordinates: 2 x 16B chunks of A and of W per thread per tile
  int sr[2], sk[2], ra[2], rb[2];
#pragma unroll
  for (int i = 0; i < 2; ++i) {
    int c = t + i * 256;                  // 0..511
    sr[i] = c >> 2;                       // 0..127
    sk[i] = (c & 3) * 8;
    ra[i] = m0 + sr[i]; if (ra[i] > M - 1) ra[i] = M - 1;   // clamp (stores guarded)
    rb[i] = n0 + sr[i]; if (rb[i] > N - 1) rb[i] = N - 1;
  }
  auto stage = [&](int buf, int kb) {
#pragma unroll
    for (int i = 0; i < 2; ++i) {
      async_ld16(&As[buf][sr[i] * TK + sk[i]], &A[(size_t)ra[i] * K + kb + sk[i]]);
      async_ld16(&Bs[buf][sr[i] * TK + sk[i]], &W[(size_t)rb[i] * K + kb + sk[i]]);
    }
  };

  floatx8 acc[4][2];
#pragma unroll
  for (int r = 0; r < 4; ++r)
#pragma unroll
    for (int c = 0; c < 2; ++c) acc[r][c] = zero8();

  const int nt = K / TK;
  stage(0, 0);
  for (int i = 0; i < nt; ++i) {
    wait_async0();               // this wave's tile-i data landed in LDS
    __syncthreads();             // everyone's tile-i landed; buf[(i+1)&1] drained
    if (i + 1 < nt) stage((i + 1) & 1, (i + 1) * TK);   // overlap with compute

    const __bf16* as = As[i & 1];
    const __bf16* bs = Bs[i & 1];
    FragAB af[4], bfr[2];
#pragma unroll
    for (int r = 0; r < 4; ++r) {
      int m = wm * 64 + r * 16 + lm;
      af[r].u[0] = *(const uint4*)&as[m * TK + koff];
      af[r].u[1] = *(const uint4*)&as[m * TK + 16 + koff];
    }
#pragma unroll
    for (int c = 0; c < 2; ++c) {
      int n = wn * 32 + c * 16 + lm;
      bfr[c].u[0] = *(const uint4*)&bs[n * TK + koff];
      bfr[c].u[1] = *(const uint4*)&bs[n * TK + 16 + koff];
    }
#pragma unroll
    for (int r = 0; r < 4; ++r)
#pragma unroll
      for (int c = 0; c < 2; ++c)
        acc[r][c] = __builtin_amdgcn_wmma_f32_16x16x32_bf16(
            false, af[r].v, false, bfr[c].v, (short)0, acc[r][c], false, false);
  }
  gemm_epilogue(acc, bias, res, outF, outB, M, N, m0, n0, wm, wn, lm, half, flags);
}

// ---------------------------------------------------------------------------
// Generic guarded WMMA GEMM (any K; used only for tiny-K adapter GEMMs)
// ---------------------------------------------------------------------------
__global__ __launch_bounds__(256) void k_gemm(
    const __bf16* __restrict__ A, const __bf16* __restrict__ W,
    const float* __restrict__ bias, const float* __restrict__ res,
    float* __restrict__ outF, __bf16* __restrict__ outB,
    int M, int N, int K, int flags)
{
  __shared__ alignas(16) __bf16 As[TM * TK];
  __shared__ alignas(16) __bf16 Bs[TN * TK];

  const int m0 = blockIdx.x * TM;
  const int n0 = blockIdx.y * TN;
  const int t  = threadIdx.x;
  const int l  = t & 31;
  const int w  = t >> 5;
  const int wm = w >> 2;
  const int wn = w & 3;
  const int lm   = l & 15;
  const int half = l >> 4;
  const int koff = half * 8;

  floatx8 acc[4][2];
#pragma unroll
  for (int r = 0; r < 4; ++r)
#pragma unroll
    for (int c = 0; c < 2; ++c) acc[r][c] = zero8();

  for (int kb = 0; kb < K; kb += TK) {
#pragma unroll
    for (int i = 0; i < 2; ++i) {
      int c  = t + i * 256;
      int r  = c >> 2;
      int kk = (c & 3) * 8;
#pragma unroll
      for (int e = 0; e < 8; ++e) {
        int gk = kb + kk + e;
        As[r * TK + kk + e] = ((m0 + r) < M && gk < K)
                                  ? A[(size_t)(m0 + r) * K + gk] : (__bf16)0.0f;
        Bs[r * TK + kk + e] = ((n0 + r) < N && gk < K)
                                  ? W[(size_t)(n0 + r) * K + gk] : (__bf16)0.0f;
      }
    }
    __syncthreads();
    FragAB af[4], bfr[2];
#pragma unroll
    for (int r = 0; r < 4; ++r) {
      int m = wm * 64 + r * 16 + lm;
      af[r].u[0] = *(const uint4*)&As[m * TK + koff];
      af[r].u[1] = *(const uint4*)&As[m * TK + 16 + koff];
    }
#pragma unroll
    for (int c = 0; c < 2; ++c) {
      int n = wn * 32 + c * 16 + lm;
      bfr[c].u[0] = *(const uint4*)&Bs[n * TK + koff];
      bfr[c].u[1] = *(const uint4*)&Bs[n * TK + 16 + koff];
    }
#pragma unroll
    for (int r = 0; r < 4; ++r)
#pragma unroll
      for (int c = 0; c < 2; ++c)
        acc[r][c] = __builtin_amdgcn_wmma_f32_16x16x32_bf16(
            false, af[r].v, false, bfr[c].v, (short)0, acc[r][c], false, false);
    __syncthreads();
  }
  gemm_epilogue(acc, bias, res, outF, outB, M, N, m0, n0, wm, wn, lm, half, flags);
}

// ---------------------------------------------------------------------------
// Fused flash attention, bf16 WMMA, online softmax, async K staging.
//   qkv: M0 x 3C bf16 (q|k|v).  out: M0 x C bf16.
//   grid.x = ceil(S/128) q-tiles, grid.y = B*H. 8 waves, 16 q-rows per wave.
// ---------------------------------------------------------------------------
__global__ __launch_bounds__(256) void k_attn(const __bf16* __restrict__ qkv,
                                              __bf16* __restrict__ out)
{
  const int bh = blockIdx.y;
  const int b = bh / Hn, h = bh % Hn;
  const int qbase = blockIdx.x * 128;
  const int t = threadIdx.x, l = t & 31, w = t >> 5;
  const int lm = l & 15, half = l >> 4, koff = half * 8;

  __shared__ alignas(16) __bf16 Kt[32 * 64];       // [key][d]
  __shared__ alignas(16) __bf16 Vt[64 * 32];       // [d][key] (transposed)
  __shared__ alignas(16) __bf16 Pt[8 * 16 * 32];   // per-wave P tiles

  // Q fragments (A-layout), loaded once from global
  int qr = qbase + w * 16 + lm; if (qr > Sn - 1) qr = Sn - 1;
  const __bf16* qp = qkv + (size_t)(b * Sn + qr) * (3 * Cn) + h * HDn;
  FragAB qf[2];
#pragma unroll
  for (int kd = 0; kd < 2; ++kd) {
    qf[kd].u[0] = *(const uint4*)(qp + kd * 32 + koff);
    qf[kd].u[1] = *(const uint4*)(qp + kd * 32 + 16 + koff);
  }

  floatx8 o[4];
#pragma unroll
  for (int dt = 0; dt < 4; ++dt) o[dt] = zero8();
  float mrow[8], lrow[8];
#pragma unroll
  for (int v = 0; v < 8; ++v) { mrow[v] = -1e30f; lrow[v] = 0.0f; }

  const int key = t >> 3;            // 0..31
  const int dcol = (t & 7) * 8;      // 0..56

  for (int kb = 0; kb < Sn; kb += 32) {
    // ---- stage K (async direct) and V (manual transpose) chunk
    {
      int gkey = kb + key; if (gkey > Sn - 1) gkey = Sn - 1;
      const __bf16* kp = qkv + (size_t)(b * Sn + gkey) * (3 * Cn) + Cn + h * HDn + dcol;
      async_ld16(&Kt[key * 64 + dcol], kp);
      const __bf16* vp = qkv + (size_t)(b * Sn + gkey) * (3 * Cn) + 2 * Cn + h * HDn + dcol;
      uint4 vv = *(const uint4*)vp;
      const __bf16* ve = (const __bf16*)&vv;
#pragma unroll
      for (int e = 0; e < 8; ++e) Vt[(dcol + e) * 32 + key] = ve[e];
    }
    wait_async0();
    __syncthreads();

    // ---- S = Q @ K^T for this wave's 16 rows x 32 keys
    floatx8 s[2]; s[0] = zero8(); s[1] = zero8();
#pragma unroll
    for (int nt = 0; nt < 2; ++nt) {
#pragma unroll
      for (int kd = 0; kd < 2; ++kd) {
        FragAB bk;
        int n = nt * 16 + lm;
        bk.u[0] = *(const uint4*)&Kt[n * 64 + kd * 32 + koff];
        bk.u[1] = *(const uint4*)&Kt[n * 64 + kd * 32 + 16 + koff];
        s[nt] = __builtin_amdgcn_wmma_f32_16x16x32_bf16(
            false, qf[kd].v, false, bk.v, (short)0, s[nt], false, false);
      }
    }
    // scale + OOB key mask
#pragma unroll
    for (int nt = 0; nt < 2; ++nt)
#pragma unroll
      for (int v = 0; v < 8; ++v) {
        int col = kb + nt * 16 + lm;
        float x = s[nt][v] * 0.125f;         // 1/sqrt(64)
        s[nt][v] = (col < Sn) ? x : -1e30f;
      }
    // ---- online softmax update (row lives on 16 lanes of one half-wave)
#pragma unroll
    for (int v = 0; v < 8; ++v) {
      float mx = fmaxf(s[0][v], s[1][v]);
#pragma unroll
      for (int msk = 1; msk < 16; msk <<= 1) mx = fmaxf(mx, __shfl_xor(mx, msk, 32));
      float mnew = fmaxf(mrow[v], mx);
      float scal = __expf(mrow[v] - mnew);
#pragma unroll
      for (int dt = 0; dt < 4; ++dt) o[dt][v] *= scal;
      float p0 = __expf(s[0][v] - mnew);
      float p1 = __expf(s[1][v] - mnew);
      s[0][v] = p0; s[1][v] = p1;
      float ls = p0 + p1;
#pragma unroll
      for (int msk = 1; msk < 16; msk <<= 1) ls += __shfl_xor(ls, msk, 32);
      lrow[v] = lrow[v] * scal + ls;
      mrow[v] = mnew;
    }
    // ---- C-layout -> A-layout for P via LDS
#pragma unroll
    for (int v = 0; v < 8; ++v) {
      int row = half * 8 + v;
      Pt[(w * 16 + row) * 32 + lm]      = (__bf16)s[0][v];
      Pt[(w * 16 + row) * 32 + 16 + lm] = (__bf16)s[1][v];
    }
    __syncthreads();
    FragAB pf;
    pf.u[0] = *(const uint4*)&Pt[(w * 16 + lm) * 32 + koff];
    pf.u[1] = *(const uint4*)&Pt[(w * 16 + lm) * 32 + 16 + koff];
    // ---- O += P @ V  (4 d-tiles of 16)
#pragma unroll
    for (int dt = 0; dt < 4; ++dt) {
      FragAB bv;
      int n = dt * 16 + lm;
      bv.u[0] = *(const uint4*)&Vt[n * 32 + koff];
      bv.u[1] = *(const uint4*)&Vt[n * 32 + 16 + koff];
      o[dt] = __builtin_amdgcn_wmma_f32_16x16x32_bf16(
          false, pf.v, false, bv.v, (short)0, o[dt], false, false);
    }
    __syncthreads();
  }

  // ---- normalize + store
#pragma unroll
  for (int dt = 0; dt < 4; ++dt)
#pragma unroll
    for (int v = 0; v < 8; ++v) {
      int q = qbase + w * 16 + half * 8 + v;
      if (q < Sn) {
        float val = o[dt][v] / lrow[v];
        out[(size_t)(b * Sn + q) * Cn + h * HDn + dt * 16 + lm] = (__bf16)val;
      }
    }
}

// ---------------------------------------------------------------------------
// LayerNorm over last dim C; one block per row; writes bf16 (and optional f32)
// ---------------------------------------------------------------------------
__global__ __launch_bounds__(256) void k_ln(const float* __restrict__ in,
                                            const float* __restrict__ g,
                                            const float* __restrict__ bta,
                                            __bf16* __restrict__ outB,
                                            float* __restrict__ outF)
{
  __shared__ float red[256];
  const int row = blockIdx.x, t = threadIdx.x;
  const float* rp = in + (size_t)row * Cn;
  float s = 0.0f;
  for (int c = t; c < Cn; c += 256) s += rp[c];
  red[t] = s; __syncthreads();
  for (int o = 128; o > 0; o >>= 1) { if (t < o) red[t] += red[t + o]; __syncthreads(); }
  float mean = red[0] / Cn; __syncthreads();
  float s2 = 0.0f;
  for (int c = t; c < Cn; c += 256) { float d = rp[c] - mean; s2 += d * d; }
  red[t] = s2; __syncthreads();
  for (int o = 128; o > 0; o >>= 1) { if (t < o) red[t] += red[t + o]; __syncthreads(); }
  float rstd = rsqrtf(red[0] / Cn + 1e-5f);
  for (int c = t; c < Cn; c += 256) {
    float y = (rp[c] - mean) * rstd * g[c] + bta[c];
    if (outB) outB[(size_t)row * Cn + c] = (__bf16)y;
    if (outF) outF[(size_t)row * Cn + c] = y;
  }
}

// ---------------------------------------------------------------------------
// Segment max/mean machinery (ordered-key float atomic max)
// ---------------------------------------------------------------------------
__device__ __forceinline__ unsigned fkey(float f) {
  unsigned u = __float_as_uint(f);
  return (u & 0x80000000u) ? ~u : (u | 0x80000000u);
}
__device__ __forceinline__ float unfkey(unsigned u) {
  return __uint_as_float((u & 0x80000000u) ? (u & 0x7FFFFFFFu) : ~u);
}

__global__ void k_seg_init(unsigned* maxk, float* sum, float* cnt, int nseg) {
  size_t n = (size_t)nseg * Cn;
  size_t i = (size_t)blockIdx.x * blockDim.x + threadIdx.x;
  size_t stride = (size_t)gridDim.x * blockDim.x;
  unsigned negk = fkey(-3e38f);
  for (size_t j = i; j < n; j += stride) { maxk[j] = negk; sum[j] = 0.0f; }
  for (size_t j = i; j < (size_t)nseg; j += stride) cnt[j] = 0.0f;
}

__global__ __launch_bounds__(256) void k_seg_accum(const float* __restrict__ feat,
                                                   const int* __restrict__ idx,
                                                   unsigned* maxk, float* sum, float* cnt)
{
  const int row = blockIdx.x, t = threadIdx.x;
  const int seg = idx[row];
  if (t == 0) atomicAdd(cnt + seg, 1.0f);
  for (int c = t; c < Cn; c += 256) {
    float v = feat[(size_t)row * Cn + c];
    atomicAdd(sum + (size_t)seg * Cn + c, v);
    atomicMax(maxk + (size_t)seg * Cn + c, fkey(v));
  }
}

__global__ void k_seg_combine(const unsigned* __restrict__ maxk,
                              const float* __restrict__ sum,
                              const float* __restrict__ cnt,
                              float* __restrict__ outT, int nseg)
{
  size_t n = (size_t)nseg * Cn;
  size_t i = (size_t)blockIdx.x * blockDim.x + threadIdx.x;
  size_t stride = (size_t)gridDim.x * blockDim.x;
  for (; i < n; i += stride) {
    int s = (int)(i / Cn);
    float c = cnt[s];
    float mx = (c > 0.0f) ? unfkey(maxk[i]) : 0.0f;
    outT[i] = mx + sum[i] / fmaxf(c, 1.0f);
  }
}

// per-column batchnorm stats over `rows` rows
__global__ __launch_bounds__(256) void k_colstats(const float* __restrict__ x, int rows,
                                                  float* __restrict__ mean,
                                                  float* __restrict__ var)
{
  __shared__ float r1[256], r2[256];
  const int c = blockIdx.x, t = threadIdx.x;
  float s = 0.0f, s2 = 0.0f;
  for (int r = t; r < rows; r += 256) { float v = x[(size_t)r * Cn + c]; s += v; s2 += v * v; }
  r1[t] = s; r2[t] = s2; __syncthreads();
  for (int o = 128; o > 0; o >>= 1) { if (t < o) { r1[t] += r1[t + o]; r2[t] += r2[t + o]; } __syncthreads(); }
  if (t == 0) { float m = r1[0] / rows; mean[c] = m; var[c] = r2[0] / rows - m * m; }
}

// BN + exact GELU elementwise
__global__ void k_bn_gelu(const float* __restrict__ x, const float* __restrict__ mean,
                          const float* __restrict__ var, const float* __restrict__ g,
                          const float* __restrict__ bta, float* __restrict__ out, int rows)
{
  size_t n = (size_t)rows * Cn;
  size_t i = (size_t)blockIdx.x * blockDim.x + threadIdx.x;
  size_t stride = (size_t)gridDim.x * blockDim.x;
  for (; i < n; i += stride) {
    int c = (int)(i % Cn);
    float y = (x[i] - mean[c]) * rsqrtf(var[c] + 1e-5f) * g[c] + bta[c];
    out[i] = 0.5f * y * (1.0f + erff(y * 0.70710678f));
  }
}

// x2 = x1 + xffn + 0.5*adapt
__global__ void k_x2(const float* x1, const float* xffn, const float* adapt, float* x2) {
  size_t n = (size_t)M0n * Cn;
  size_t i = (size_t)blockIdx.x * blockDim.x + threadIdx.x;
  size_t stride = (size_t)gridDim.x * blockDim.x;
  for (; i < n; i += stride) x2[i] = x1[i] + xffn[i] + 0.5f * adapt[i];
}

// extract tokens 1..G into contiguous feat (f32 + bf16)
__global__ void k_feat(const float* x2, float* featF, __bf16* featB) {
  size_t n = (size_t)BGn * Cn;
  size_t i = (size_t)blockIdx.x * blockDim.x + threadIdx.x;
  size_t stride = (size_t)gridDim.x * blockDim.x;
  for (; i < n; i += stride) {
    int bg = (int)(i / Cn), c = (int)(i % Cn);
    int b = bg >> 9, g = bg & 511;
    float v = x2[((size_t)(b * Sn + 1 + g)) * Cn + c];
    featF[i] = v; featB[i] = (__bf16)v;
  }
}

// cls token passthrough
__global__ void k_cls(const float* x2, float* out) {
  int n = Bn * Cn;
  int i = blockIdx.x * blockDim.x + threadIdx.x;
  if (i < n) {
    int b = i / Cn, c = i % Cn;
    out[((size_t)b * Sn) * Cn + c] = x2[((size_t)b * Sn) * Cn + c];
  }
}

// ---------------------------------------------------------------------------
// Mini windowed attention (head dim 2) for one view; grid.x = 128 windows.
// ---------------------------------------------------------------------------
__global__ __launch_bounds__(128) void k_view_attn(const float* __restrict__ qkv1,
                                                   const unsigned char* __restrict__ mask,
                                                   __bf16* __restrict__ o1)
{
  __shared__ float kk[N1n][MIDn];
  __shared__ float vv[N1n][MIDn];
  const int wdw = blockIdx.x, t = threadIdx.x;
  const int row = wdw * N1n + t;
  const float* rp = qkv1 + (size_t)row * (3 * MIDn);
  float q[MIDn];
#pragma unroll
  for (int i = 0; i < MIDn; ++i) {
    q[i] = rp[i];
    kk[t][i] = rp[MIDn + i];
    vv[t][i] = rp[2 * MIDn + i];
  }
  __syncthreads();
  const unsigned char* mrow = mask + ((size_t)wdw * N1n + t) * N1n;
  const float sc = 0.2886751346f;    // 12^-0.5
  float o[MIDn];
#pragma unroll
  for (int h = 0; h < H1n; ++h) {
    float q0 = q[h * 2] * sc, q1 = q[h * 2 + 1] * sc;
    float mx = -1e30f;
    for (int j = 0; j < N1n; ++j) {
      float lg = q0 * kk[j][h * 2] + q1 * kk[j][h * 2 + 1];
      if (mrow[j]) lg -= 100000.0f;
      mx = fmaxf(mx, lg);
    }
    float sum = 0.0f, a0 = 0.0f, a1 = 0.0f;
    for (int j = 0; j < N1n; ++j) {
      float lg = q0 * kk[j][h * 2] + q1 * kk[j][h * 2 + 1];
      if (mrow[j]) lg -= 100000.0f;
      float p = __expf(lg - mx);
      sum += p; a0 += p * vv[j][h * 2]; a1 += p * vv[j][h * 2 + 1];
    }
    o[h * 2] = a0 / sum; o[h * 2 + 1] = a1 / sum;
  }
#pragma unroll
  for (int i = 0; i < MIDn; ++i) o1[(size_t)row * MIDn + i] = (__bf16)o[i];
}

// ---------------------------------------------------------------------------
// Final cosine-similarity fusion; one block per token (b,g)
// ---------------------------------------------------------------------------
__device__ __forceinline__ float blockReduceSum(float v, float* red) {
  int t = threadIdx.x;
  red[t] = v; __syncthreads();
  for (int o = 128; o > 0; o >>= 1) { if (t < o) red[t] += red[t + o]; __syncthreads(); }
  float r = red[0]; __syncthreads();
  return r;
}

__global__ __launch_bounds__(256) void k_fuse(const float* __restrict__ feat,
                                              const float* __restrict__ x3dtab,
                                              const float* __restrict__ postab,
                                              const int* __restrict__ cluster,
                                              const int* __restrict__ fgi,
                                              float* __restrict__ out)
{
  __shared__ float red[256];
  __shared__ float sims[NVn];
  const int bg = blockIdx.x, t = threadIdx.x;
  const int b = bg >> 9, g = bg & 511;
  const float* x3 = x3dtab + (size_t)cluster[bg] * Cn;

  float sx3 = 0.0f;
  for (int c = t; c < Cn; c += 256) { float v = x3[c]; sx3 += v * v; }
  float nx3 = fmaxf(sqrtf(blockReduceSum(sx3, red)), 1e-8f);

  for (int n = 0; n < NVn; ++n) {
    const float* p = postab + ((size_t)n * D1n + fgi[(size_t)n * BGn + bg]) * Cn;
    float d = 0.0f, qq = 0.0f;
    for (int c = t; c < Cn; c += 256) { float pv = p[c]; d += pv * x3[c]; qq += pv * pv; }
    d = blockReduceSum(d, red);
    qq = blockReduceSum(qq, red);
    if (t == 0) sims[n] = (d / (fmaxf(sqrtf(qq), 1e-8f) * nx3) + 1.0f) * 0.5f;
    __syncthreads();
  }
  if (t == 0) {
    float s = 0.0f;
    for (int n = 0; n < NVn; ++n) s += sims[n];
    for (int n = 0; n < NVn; ++n) sims[n] /= s;
  }
  __syncthreads();
  for (int c = t; c < Cn; c += 256) {
    float sup = 0.0f;
    for (int n = 0; n < NVn; ++n)
      sup += sims[n] * postab[((size_t)n * D1n + fgi[(size_t)n * BGn + bg]) * Cn + c];
    out[((size_t)(b * Sn + 1 + g)) * Cn + c] = feat[(size_t)bg * Cn + c] + 0.3f * sup;
  }
}

// ---------------------------------------------------------------------------
// Host side
// ---------------------------------------------------------------------------
static inline size_t alignUp(size_t x) { return (x + 255) & ~(size_t)255; }
static inline int cdiv(int a, int b) { return (a + b - 1) / b; }

extern "C" void kernel_launch(void* const* d_in, const int* in_sizes, int n_in,
                              void* d_out, int out_size, void* d_ws, size_t ws_size,
                              hipStream_t stream) {
  (void)in_sizes; (void)n_in; (void)out_size; (void)ws_size;
  const float* x        = (const float*)d_in[0];
  const float* ln1_g    = (const float*)d_in[1];
  const float* ln1_b    = (const float*)d_in[2];
  const float* ln2_g    = (const float*)d_in[3];
  const float* ln2_b    = (const float*)d_in[4];
  const float* Wqkv     = (const float*)d_in[5];
  const float* bqkv     = (const float*)d_in[6];
  const float* Wo       = (const float*)d_in[7];
  const float* bo       = (const float*)d_in[8];
  const float* Wfc      = (const float*)d_in[9];
  const float* bfc      = (const float*)d_in[10];
  const float* Wproj    = (const float*)d_in[11];
  const float* bproj    = (const float*)d_in[12];
  const float* Wa1      = (const float*)d_in[13];
  const float* ba1      = (const float*)d_in[14];
  const float* Wa2      = (const float*)d_in[15];
  const float* ba2      = (const float*)d_in[16];
  const float* bn3d_g   = (const float*)d_in[17];
  const float* bn3d_b   = (const float*)d_in[18];
  const float* bn1d_g   = (const float*)d_in[19];
  const float* bn1d_b   = (const float*)d_in[20];
  const float* n3_g     = (const float*)d_in[21];
  const float* n3_b     = (const float*)d_in[22];
  const float* a1_qkv_w = (const float*)d_in[23];
  const float* a1_qkv_b = (const float*)d_in[24];
  const float* a1_proj_w= (const float*)d_in[25];
  const float* a1_proj_b= (const float*)d_in[26];
  const int*   cluster  = (const int*)d_in[27];
  const int*   fgi      = (const int*)d_in[28];
  const unsigned char* mask = (const unsigned char*)d_in[29];
  float* out = (float*)d_out;

  // ---- bump allocator over d_ws
  char* wp = (char*)d_ws;
  auto alloc = [&](size_t bytes) -> void* { void* p = wp; wp += alignUp(bytes); return p; };

  __bf16* wqkv_bf   = (__bf16*)alloc((size_t)3 * Cn * Cn * 2);
  __bf16* wo_bf     = (__bf16*)alloc((size_t)Cn * Cn * 2);
  __bf16* wfc_bf    = (__bf16*)alloc((size_t)FFn * Cn * 2);
  __bf16* wproj_bf  = (__bf16*)alloc((size_t)Cn * FFn * 2);
  __bf16* wa1_bf    = (__bf16*)alloc((size_t)ADn * Cn * 2);
  __bf16* wa2_bf    = (__bf16*)alloc((size_t)Cn * ADn * 2);
  __bf16* vqkvw_bf  = (__bf16*)alloc((size_t)NVn * 3 * MIDn * Cn * 2);
  __bf16* vprojw_bf = (__bf16*)alloc((size_t)NVn * Cn * MIDn * 2);
  __bf16* lnbuf     = (__bf16*)alloc((size_t)M0n * Cn * 2);     // LN outputs (reused)
  __bf16* bigbuf    = (__bf16*)alloc((size_t)M0n * FFn * 2);    // qkv then h
  __bf16* abuf      = (__bf16*)alloc((size_t)M0n * Cn * 2);     // attn_out then xffn_bf
  float*  x1        = (float*)alloc((size_t)M0n * Cn * 4);
  float*  xffn      = (float*)alloc((size_t)M0n * Cn * 4);
  float*  adapt     = (float*)alloc((size_t)M0n * Cn * 4);
  float*  x2        = (float*)alloc((size_t)M0n * Cn * 4);
  float*  featF     = (float*)alloc((size_t)BGn * Cn * 4);
  __bf16* featB     = (__bf16*)alloc((size_t)BGn * Cn * 2);
  __bf16* a_bf      = (__bf16*)alloc((size_t)M0n * ADn * 2);
  unsigned* maxk    = (unsigned*)alloc((size_t)D1n * Cn * 4);
  float*  ssum      = (float*)alloc((size_t)D1n * Cn * 4);
  float*  scnt      = (float*)alloc((size_t)D1n * 4);
  float*  ttab      = (float*)alloc((size_t)D1n * Cn * 4);
  float*  meanv     = (float*)alloc((size_t)Cn * 4);
  float*  varv      = (float*)alloc((size_t)Cn * 4);
  float*  x3dtab    = (float*)alloc((size_t)NCLn * Cn * 4);
  float*  qkv1f     = (float*)alloc((size_t)BGn * 3 * MIDn * 4);
  __bf16* o1_bf     = (__bf16*)alloc((size_t)BGn * MIDn * 2);
  float*  flatb     = (float*)alloc((size_t)BGn * Cn * 4);
  float*  postab    = (float*)alloc((size_t)NVn * D1n * Cn * 4);

  auto cvt = [&](const float* s, __bf16* d, size_t n) {
    int blocks = (int)((n + 255) / 256); if (blocks > 4096) blocks = 4096;
    k_cvt_bf16<<<blocks, 256, 0, stream>>>(s, d, n);
  };
  auto gemm = [&](const __bf16* A, const __bf16* W, const float* bias, const float* res,
                  float* oF, __bf16* oB, int M, int N, int K, int flags) {
    dim3 grid(cdiv(M, TM), cdiv(N, TN));
    if ((K % TK) == 0)
      k_gemm_fast<<<grid, 256, 0, stream>>>(A, W, bias, res, oF, oB, M, N, K, flags);
    else
      k_gemm<<<grid, 256, 0, stream>>>(A, W, bias, res, oF, oB, M, N, K, flags);
  };

  // ---- weight conversions
  cvt(Wqkv,     wqkv_bf,   (size_t)3 * Cn * Cn);
  cvt(Wo,       wo_bf,     (size_t)Cn * Cn);
  cvt(Wfc,      wfc_bf,    (size_t)FFn * Cn);
  cvt(Wproj,    wproj_bf,  (size_t)Cn * FFn);
  cvt(Wa1,      wa1_bf,    (size_t)ADn * Cn);
  cvt(Wa2,      wa2_bf,    (size_t)Cn * ADn);
  cvt(a1_qkv_w, vqkvw_bf,  (size_t)NVn * 3 * MIDn * Cn);
  cvt(a1_proj_w,vprojw_bf, (size_t)NVn * Cn * MIDn);

  // ---- transformer block
  k_ln<<<M0n, 256, 0, stream>>>(x, ln1_g, ln1_b, lnbuf, nullptr);
  gemm(lnbuf, wqkv_bf, bqkv, nullptr, nullptr, bigbuf, M0n, 3 * Cn, Cn, EP_BF16);
  {
    dim3 grid(cdiv(Sn, 128), Bn * Hn);
    k_attn<<<grid, 256, 0, stream>>>(bigbuf, abuf);
  }
  gemm(abuf, wo_bf, bo, x, x1, nullptr, M0n, Cn, Cn, EP_F32 | EP_RES);
  k_ln<<<M0n, 256, 0, stream>>>(x1, ln2_g, ln2_b, lnbuf, nullptr);
  gemm(lnbuf, wfc_bf, bfc, nullptr, nullptr, bigbuf, M0n, FFn, Cn, EP_BF16 | EP_GELU);
  gemm(bigbuf, wproj_bf, bproj, nullptr, xffn, abuf, M0n, Cn, FFn, EP_F32 | EP_BF16);
  gemm(abuf, wa1_bf, ba1, nullptr, nullptr, a_bf, M0n, ADn, Cn, EP_BF16 | EP_GELU);
  gemm(a_bf, wa2_bf, ba2, nullptr, adapt, nullptr, M0n, Cn, ADn, EP_F32);
  k_x2<<<4096, 256, 0, stream>>>(x1, xffn, adapt, x2);
  k_feat<<<4096, 256, 0, stream>>>(x2, featF, featB);

  // ---- 3D path: cluster seg max/mean + BN-GELU
  k_seg_init<<<4096, 256, 0, stream>>>(maxk, ssum, scnt, NCLn);
  k_seg_accum<<<BGn, 256, 0, stream>>>(featF, cluster, maxk, ssum, scnt);
  k_seg_combine<<<4096, 256, 0, stream>>>(maxk, ssum, scnt, ttab, NCLn);
  k_colstats<<<Cn, 256, 0, stream>>>(ttab, NCLn, meanv, varv);
  k_bn_gelu<<<4096, 256, 0, stream>>>(ttab, meanv, varv, bn3d_g, bn3d_b, x3dtab, NCLn);

  // ---- view branches
  for (int n = 0; n < NVn; ++n) {
    k_ln<<<BGn, 256, 0, stream>>>(featF, n3_g + (size_t)n * Cn, n3_b + (size_t)n * Cn,
                                  lnbuf, nullptr);
    gemm(lnbuf, vqkvw_bf + (size_t)n * 3 * MIDn * Cn, a1_qkv_b + (size_t)n * 3 * MIDn,
         nullptr, qkv1f, nullptr, BGn, 3 * MIDn, Cn, EP_F32);
    k_view_attn<<<BGn / N1n, N1n, 0, stream>>>(
        qkv1f, mask + (size_t)n * (BGn / N1n) * N1n * N1n, o1_bf);
    gemm(o1_bf, vprojw_bf + (size_t)n * Cn * MIDn, a1_proj_b + (size_t)n * Cn,
         featF, flatb, nullptr, BGn, Cn, MIDn, EP_F32 | EP_RES);
    k_seg_init<<<4096, 256, 0, stream>>>(maxk, ssum, scnt, D1n);
    k_seg_accum<<<BGn, 256, 0, stream>>>(flatb, fgi + (size_t)n * BGn, maxk, ssum, scnt);
    k_seg_combine<<<4096, 256, 0, stream>>>(maxk, ssum, scnt, ttab, D1n);
    k_colstats<<<Cn, 256, 0, stream>>>(ttab, D1n, meanv, varv);
    k_bn_gelu<<<4096, 256, 0, stream>>>(ttab, meanv, varv, bn1d_g + (size_t)n * Cn,
                                        bn1d_b + (size_t)n * Cn,
                                        postab + (size_t)n * D1n * Cn, D1n);
  }

  // ---- fusion + output
  k_cls<<<cdiv(Bn * Cn, 256), 256, 0, stream>>>(x2, out);
  k_fuse<<<BGn, 256, 0, stream>>>(featF, x3dtab, postab, cluster, fgi, out);
}
